// GptLanguageModel_86930138071258
// MI455X (gfx1250) — matrix operand order repaired and checked by reading, not compiled, not run on hardware
//
#include <hip/hip_runtime.h>
#include <stddef.h>
#include <stdint.h>

// ---------------------------------------------------------------------------
// GPT forward for gfx1250 (MI455X). All matmuls via V_WMMA_F32_16X16X32_BF16;
// global->LDS copies of bf16 tiles use GLOBAL_LOAD_ASYNC_TO_LDS_B128
// (ASYNCcnt / s_wait_asynccnt), B panels are reg-staged + converted to bf16.
// B=2 T=1024 E=1024 H=16 G=4 NKV=4 D=64 L=8 V=32000
// ---------------------------------------------------------------------------

typedef __bf16  v16bf __attribute__((ext_vector_type(16)));
typedef __bf16  v8bf  __attribute__((ext_vector_type(8)));
typedef float   v8f   __attribute__((ext_vector_type(8)));

#define WMMA_BF16(a, b, c) \
  __builtin_amdgcn_wmma_f32_16x16x32_bf16(false, (a), false, (b), (short)0, (c), false, false)

__device__ __forceinline__ v8bf ld8(const __bf16* p) { return *(const v8bf*)p; }
__device__ __forceinline__ v16bf mk16(v8bf lo, v8bf hi) {
  return __builtin_shufflevector(lo, hi, 0,1,2,3,4,5,6,7,8,9,10,11,12,13,14,15);
}

// Async global->LDS copy of 16 bytes (one b128 per lane). lds_off is the
// 32-bit LDS byte offset (low 32 bits of a generic pointer into __shared__).
__device__ __forceinline__ void async_cp16(void* lds_ptr, const void* gptr) {
  uint32_t loff = (uint32_t)(uintptr_t)lds_ptr;
  asm volatile("global_load_async_to_lds_b128 %0, %1, off"
               :: "v"(loff), "v"(gptr) : "memory");
}
__device__ __forceinline__ void wait_async0() {
  asm volatile("s_wait_asynccnt 0x0" ::: "memory");
}

// ---------------------------------------------------------------------------
// x[b,t,:] = tok_emb[idx[b,t],:] + pos_emb[t,:]
// ---------------------------------------------------------------------------
__global__ __launch_bounds__(256) void embed_kernel(
    const int* __restrict__ idx, const float* __restrict__ tok,
    const float* __restrict__ pos, float* __restrict__ x)
{
  const int bt = blockIdx.x;
  const int t  = bt & 1023;
  const int tk = idx[bt];
  const int e  = threadIdx.x * 4;
  float4 a = *(const float4*)(tok + (size_t)tk * 1024 + e);
  float4 p = *(const float4*)(pos + (size_t)t  * 1024 + e);
  float4 r; r.x = a.x + p.x; r.y = a.y + p.y; r.z = a.z + p.z; r.w = a.w + p.w;
  *(float4*)(x + (size_t)bt * 1024 + e) = r;
}

// ---------------------------------------------------------------------------
// LayerNorm over E=1024. One block per row, f32 + bf16 outputs (optional).
// ---------------------------------------------------------------------------
__global__ __launch_bounds__(256) void ln_kernel(
    const float* __restrict__ x, const float* __restrict__ w,
    float* __restrict__ yf, __bf16* __restrict__ yb)
{
  __shared__ float red[256];
  const int row = blockIdx.x, tid = threadIdx.x;
  const float* xr = x + (size_t)row * 1024;
  float4 v = *(const float4*)(xr + tid * 4);

  red[tid] = v.x + v.y + v.z + v.w;
  __syncthreads();
  for (int off = 128; off > 0; off >>= 1) {
    if (tid < off) red[tid] += red[tid + off];
    __syncthreads();
  }
  const float mu = red[0] * (1.0f / 1024.0f);
  __syncthreads();

  const float d0 = v.x - mu, d1 = v.y - mu, d2 = v.z - mu, d3 = v.w - mu;
  red[tid] = d0 * d0 + d1 * d1 + d2 * d2 + d3 * d3;
  __syncthreads();
  for (int off = 128; off > 0; off >>= 1) {
    if (tid < off) red[tid] += red[tid + off];
    __syncthreads();
  }
  const float rstd = rsqrtf(red[0] * (1.0f / 1024.0f) + 1e-5f);

  float4 wv = *(const float4*)(w + tid * 4);
  const float o0 = d0 * rstd * wv.x, o1 = d1 * rstd * wv.y;
  const float o2 = d2 * rstd * wv.z, o3 = d3 * rstd * wv.w;
  if (yf) {
    float4 o; o.x = o0; o.y = o1; o.z = o2; o.w = o3;
    *(float4*)(yf + (size_t)row * 1024 + tid * 4) = o;
  }
  if (yb) {
    __bf16* p = yb + (size_t)row * 1024 + tid * 4;
    p[0] = (__bf16)o0; p[1] = (__bf16)o1; p[2] = (__bf16)o2; p[3] = (__bf16)o3;
  }
}

// ---------------------------------------------------------------------------
// WMMA GEMM: C[M,N] = A_bf16[M,K] @ B_f32[K,N]  (+res, relu, f32/bf16 out)
// 128x128 block tile, k-step 32, double-buffered LDS, software pipeline:
//   A tile    : GLOBAL_LOAD_ASYNC_TO_LDS_B128 (ASYNCcnt)
//   B tile    : coalesced f32 loads -> regs -> bf16 pack -> 2x ds_store_b128
// M,N multiples of 128; K multiple of 32.
// ---------------------------------------------------------------------------
template<bool RES, bool RELU, bool WF32, bool WBF16>
__global__ __launch_bounds__(256) void gemm_bf16_kernel(
    const __bf16* __restrict__ A, const float* __restrict__ B,
    const float* __restrict__ res, float* __restrict__ Cf,
    __bf16* __restrict__ Cb, int M, int N, int K)
{
  __shared__ __align__(16) __bf16 ldsA[2][128][40];   // [buf][m][k], pad 8
  __shared__ __align__(16) __bf16 ldsBt[2][128][40];  // [buf][n][k], pad 8

  const int tid  = threadIdx.x;
  const int lane = tid & 31, w = tid >> 5;
  const int hi   = lane >> 4, ln16 = lane & 15;
  const int wr   = w >> 1, wc = w & 1;                 // 4x2 wave grid
  const int m0   = blockIdx.y * 128, n0 = blockIdx.x * 128;

  const int ar = tid >> 1, acg = (tid & 1) * 16;       // A: row, 16-elem group
  const int bn = tid & 127, bh = (tid >> 7) * 16;      // B: col n, k-half

  float br[16];
  v8f c[2][4] = {};

  auto issueA = [&](int buf, int k0) {
    const __bf16* g = A + (size_t)(m0 + ar) * K + k0 + acg;
    async_cp16(&ldsA[buf][ar][acg],     g);
    async_cp16(&ldsA[buf][ar][acg + 8], g + 8);
  };
  auto loadB = [&](int k0) {
#pragma unroll
    for (int i = 0; i < 16; ++i)
      br[i] = B[(size_t)(k0 + bh + i) * N + n0 + bn];  // coalesced over bn
  };
  auto storeB = [&](int buf) {
    v8bf lo, hv;
#pragma unroll
    for (int i = 0; i < 8; ++i) { lo[i] = (__bf16)br[i]; hv[i] = (__bf16)br[8 + i]; }
    *(v8bf*)&ldsBt[buf][bn][bh]     = lo;
    *(v8bf*)&ldsBt[buf][bn][bh + 8] = hv;
  };

  // Prologue: stage tile 0.
  issueA(0, 0);
  loadB(0);
  storeB(0);
  wait_async0();
  __syncthreads();

  int cur = 0;
  for (int k0 = 0; k0 < K; k0 += 32) {
    const bool more = (k0 + 32) < K;
    if (more) {
      issueA(cur ^ 1, k0 + 32);                        // async, other buffer
      loadB(k0 + 32);                                  // into registers
      if (k0 + 64 < K)
        __builtin_prefetch(B + (size_t)(k0 + 64 + bh) * N + n0 + bn, 0, 1);
    }

    // Compute current tile.
    v16bf a[2], b[4];
#pragma unroll
    for (int rt = 0; rt < 2; ++rt) {
      const __bf16* p = &ldsA[cur][wr * 32 + rt * 16 + ln16][hi * 8];
      a[rt] = mk16(ld8(p), ld8(p + 16));               // CDNA5 16-bit A layout
    }
#pragma unroll
    for (int ct = 0; ct < 4; ++ct) {
      const __bf16* p = &ldsBt[cur][wc * 64 + ct * 16 + ln16][hi * 8];
      b[ct] = mk16(ld8(p), ld8(p + 16));               // CDNA5 16-bit B layout
    }
#pragma unroll
    for (int rt = 0; rt < 2; ++rt)
#pragma unroll
      for (int ct = 0; ct < 4; ++ct)
        c[rt][ct] = WMMA_BF16(a[rt], b[ct], c[rt][ct]);

    if (more) {
      storeB(cur ^ 1);                                 // waits loadcnt here
      wait_async0();                                   // A tile landed in LDS
    }
    __syncthreads();
    cur ^= 1;
  }

  // Epilogue (C layout: lane -> col = ln16; vgpr j -> row = j + 8*hi)
#pragma unroll
  for (int rt = 0; rt < 2; ++rt)
#pragma unroll
    for (int ct = 0; ct < 4; ++ct)
#pragma unroll
      for (int j = 0; j < 8; ++j) {
        const int gr = m0 + wr * 32 + rt * 16 + j + hi * 8;
        const int gc = n0 + wc * 64 + ct * 16 + ln16;
        float v = c[rt][ct][j];
        if (RES)  v += res[(size_t)gr * N + gc];
        if (RELU) v = v > 0.0f ? v : 0.0f;
        if (WF32)  Cf[(size_t)gr * N + gc] = v;
        if (WBF16) Cb[(size_t)gr * N + gc] = (__bf16)v;
      }
}

// ---------------------------------------------------------------------------
// Flash attention. grid = (T/128, B*G*NKV). One wave = 16 query rows.
// K tile copied via async global->LDS; V tile reg-staged + transposed.
// ---------------------------------------------------------------------------
__global__ __launch_bounds__(256) void attn_kernel(
    const __bf16* __restrict__ q, const __bf16* __restrict__ k,
    const __bf16* __restrict__ v, __bf16* __restrict__ o)
{
  __shared__ __align__(16) __bf16 ldsK[128][72];    // [key][d]
  __shared__ __align__(16) __bf16 ldsVt[64][136];   // [d][key]
  __shared__ __align__(16) __bf16 ldsP[128][136];   // per-wave 16-row slices

  const int tid  = threadIdx.x;
  const int lane = tid & 31, w = tid >> 5;
  const int hi   = lane >> 4, ln16 = lane & 15;
  const int qblk = blockIdx.x;             // 0..7
  const int hz   = blockIdx.y;             // 0..31
  const int b    = hz >> 4, gn = hz & 15, n = gn & 3;

  // Q fragments for this wave's 16 rows (D=64 -> 2 k-steps), kept in regs.
  v16bf aq[2];
  {
    const int t = qblk * 128 + w * 16 + ln16;
    const __bf16* qrow = q + ((size_t)(b * 1024 + t)) * 1024 + gn * 64;
#pragma unroll
    for (int ks = 0; ks < 2; ++ks) {
      const __bf16* p = qrow + ks * 32 + hi * 8;
      aq[ks] = mk16(*(const v8bf*)p, *(const v8bf*)(p + 16));
    }
  }

  float mrow[8], lrow[8];
  v8f oacc[4] = {};
  for (int j = 0; j < 8; ++j) { mrow[j] = -1e30f; lrow[j] = 0.0f; }

  const int tkey = tid >> 1, tdg = (tid & 1) * 32;   // tile copy roles

  for (int kblk = 0; kblk <= qblk; ++kblk) {
    // K tile 128x64 bf16: async global->LDS (4x b128 per thread).
    {
      const __bf16* kr = k + ((size_t)(b * 1024 + kblk * 128 + tkey)) * 256 + n * 64 + tdg;
#pragma unroll
      for (int cc = 0; cc < 4; ++cc)
        async_cp16(&ldsK[tkey][tdg + cc * 8], kr + cc * 8);
    }
    // V tile: reg-stage 32 bf16 per thread (4x b128), transpose into LDS.
    {
      const __bf16* vr = v + ((size_t)(b * 1024 + kblk * 128 + tkey)) * 256 + n * 64 + tdg;
      v8bf r0 = ((const v8bf*)vr)[0], r1 = ((const v8bf*)vr)[1];
      v8bf r2 = ((const v8bf*)vr)[2], r3 = ((const v8bf*)vr)[3];
#pragma unroll
      for (int i = 0; i < 8; ++i) {
        ldsVt[tdg + i][tkey]      = r0[i];
        ldsVt[tdg + 8 + i][tkey]  = r1[i];
        ldsVt[tdg + 16 + i][tkey] = r2[i];
        ldsVt[tdg + 24 + i][tkey] = r3[i];
      }
    }
    wait_async0();
    __syncthreads();

    // S = Q @ K^T : 8 tiles of 16x16, k-dim = D = 64 (2 steps).
    v8f s[8] = {};
#pragma unroll
    for (int ks = 0; ks < 2; ++ks)
#pragma unroll
      for (int t = 0; t < 8; ++t) {
        const __bf16* p = &ldsK[t * 16 + ln16][ks * 32 + hi * 8];
        v16bf bf = mk16(ld8(p), ld8(p + 16));
        s[t] = WMMA_BF16(aq[ks], bf, s[t]);
      }

    // Scale + causal mask + per-row max.
    float rmax[8];
    for (int j = 0; j < 8; ++j) rmax[j] = -1e30f;
#pragma unroll
    for (int t = 0; t < 8; ++t) {
      const int key = kblk * 128 + t * 16 + ln16;
#pragma unroll
      for (int j = 0; j < 8; ++j) {
        const int qg = qblk * 128 + w * 16 + j + hi * 8;
        float sv = s[t][j] * 0.125f;             // D^-0.5
        if (key > qg) sv = -1e30f;
        s[t][j] = sv;
        rmax[j] = fmaxf(rmax[j], sv);
      }
    }
#pragma unroll
    for (int j = 0; j < 8; ++j)
      for (int m = 1; m < 16; m <<= 1)
        rmax[j] = fmaxf(rmax[j], __shfl_xor(rmax[j], m, 32));

    float alpha[8];
#pragma unroll
    for (int j = 0; j < 8; ++j) {
      const float mn = fmaxf(mrow[j], rmax[j]);
      alpha[j] = __expf(mrow[j] - mn);
      mrow[j] = mn;
    }

    // P = exp(S - m), stash bf16 in this wave's private LDS slice.
    float psum[8] = {};
#pragma unroll
    for (int t = 0; t < 8; ++t)
#pragma unroll
      for (int j = 0; j < 8; ++j) {
        const float p = __expf(s[t][j] - mrow[j]);
        psum[j] += p;
        ldsP[w * 16 + j + hi * 8][t * 16 + ln16] = (__bf16)p;
      }
#pragma unroll
    for (int j = 0; j < 8; ++j) {
      for (int m = 1; m < 16; m <<= 1) psum[j] += __shfl_xor(psum[j], m, 32);
      lrow[j] = lrow[j] * alpha[j] + psum[j];
    }
#pragma unroll
    for (int dt = 0; dt < 4; ++dt)
#pragma unroll
      for (int j = 0; j < 8; ++j) oacc[dt][j] *= alpha[j];

    // O += P @ V  (k-dim = 128 keys, 4 steps; per-wave LDS, in-order DS ops)
#pragma unroll
    for (int ks = 0; ks < 4; ++ks) {
      const __bf16* pp = &ldsP[w * 16 + ln16][ks * 32 + hi * 8];
      v16bf ap = mk16(ld8(pp), ld8(pp + 16));
#pragma unroll
      for (int dt = 0; dt < 4; ++dt) {
        const __bf16* vp = &ldsVt[dt * 16 + ln16][ks * 32 + hi * 8];
        v16bf bv = mk16(ld8(vp), ld8(vp + 16));
        oacc[dt] = WMMA_BF16(ap, bv, oacc[dt]);
      }
    }
    __syncthreads();
  }

  // Normalize and store o[b,t, gn*64 + d] bf16.
#pragma unroll
  for (int dt = 0; dt < 4; ++dt)
#pragma unroll
    for (int j = 0; j < 8; ++j) {
      const int t   = qblk * 128 + w * 16 + j + hi * 8;
      const int col = gn * 64 + dt * 16 + ln16;
      o[((size_t)(b * 1024 + t)) * 1024 + col] = (__bf16)(oacc[dt][j] / lrow[j]);
    }
}

// ---------------------------------------------------------------------------
// Host-side orchestration
// ---------------------------------------------------------------------------
extern "C" void kernel_launch(void* const* d_in, const int* in_sizes, int n_in,
                              void* d_out, int out_size, void* d_ws, size_t ws_size,
                              hipStream_t stream) {
  const int*   idx = (const int*)  d_in[0];
  const float* tok = (const float*)d_in[1];
  const float* pos = (const float*)d_in[2];
  const float* qp  = (const float*)d_in[3];   // [L,E,1024]
  const float* kvp = (const float*)d_in[4];   // [L,2,E,256]
  const float* op  = (const float*)d_in[5];   // [L,E,E]
  const float* f1  = (const float*)d_in[6];   // [L,E,4096]
  const float* f2  = (const float*)d_in[7];   // [L,4096,E]
  const float* sc  = (const float*)d_in[8];   // [L,2,E]
  const float* osc = (const float*)d_in[9];   // [E]
  const float* lmh = (const float*)d_in[10];  // [E,V]
  float* out = (float*)d_out;

  // Workspace carve (62 MiB total).
  char* ws = (char*)d_ws;
  float*  x   = (float*) ws; ws += (size_t)8  << 20;   // residual / layer io
  float*  hf  = (float*) ws; ws += (size_t)8  << 20;   // LN1 out f32
  __bf16* hb  = (__bf16*)ws; ws += (size_t)4  << 20;   // LN1 out bf16
  float*  h2f = (float*) ws; ws += (size_t)8  << 20;   // LN2 out f32
  __bf16* h2b = (__bf16*)ws; ws += (size_t)4  << 20;   // LN2 out bf16
  __bf16* qb  = (__bf16*)ws; ws += (size_t)4  << 20;   // q   [2048,1024]
  __bf16* kb  = (__bf16*)ws; ws += (size_t)1  << 20;   // k   [2048,256]
  __bf16* vb  = (__bf16*)ws; ws += (size_t)1  << 20;   // v   [2048,256]
  __bf16* ob  = (__bf16*)ws; ws += (size_t)4  << 20;   // attn out [2048,1024]
  __bf16* mb  = (__bf16*)ws; ws += (size_t)16 << 20;   // mlp hidden [2048,4096]
  __bf16* xfb = (__bf16*)ws; ws += (size_t)4  << 20;   // final LN bf16

  const dim3 blk(256);
  embed_kernel<<<2048, blk, 0, stream>>>(idx, tok, pos, x);

  for (int l = 0; l < 8; ++l) {
    const float* s0 = sc + (size_t)(l * 2 + 0) * 1024;
    const float* s1 = sc + (size_t)(l * 2 + 1) * 1024;
    const float* wq = qp  + (size_t)l * 1024 * 1024;
    const float* wk = kvp + (size_t)l * 2 * 1024 * 256;
    const float* wv = wk + (size_t)1024 * 256;
    const float* wo = op + (size_t)l * 1024 * 1024;
    const float* w1 = f1 + (size_t)l * 1024 * 4096;
    const float* w2 = f2 + (size_t)l * 4096 * 1024;

    // h = LN(x) (f32 for residual, bf16 for matmuls)
    ln_kernel<<<2048, blk, 0, stream>>>(x, s0, hf, hb);
    // q/k/v projections
    gemm_bf16_kernel<false,false,false,true><<<dim3(8, 16), blk, 0, stream>>>(
        hb, wq, nullptr, nullptr, qb, 2048, 1024, 1024);
    gemm_bf16_kernel<false,false,false,true><<<dim3(2, 16), blk, 0, stream>>>(
        hb, wk, nullptr, nullptr, kb, 2048, 256, 1024);
    gemm_bf16_kernel<false,false,false,true><<<dim3(2, 16), blk, 0, stream>>>(
        hb, wv, nullptr, nullptr, vb, 2048, 256, 1024);
    // attention
    attn_kernel<<<dim3(8, 32), blk, 0, stream>>>(qb, kb, vb, ob);
    // r = h + o @ Wout  (reuse x as r)
    gemm_bf16_kernel<true,false,true,false><<<dim3(8, 16), blk, 0, stream>>>(
        ob, wo, hf, x, nullptr, 2048, 1024, 1024);
    // h2 = LN(r)
    ln_kernel<<<2048, blk, 0, stream>>>(x, s1, h2f, h2b);
    // mlp hidden = relu(h2 @ fc1)
    gemm_bf16_kernel<false,true,false,true><<<dim3(32, 16), blk, 0, stream>>>(
        h2b, w1, nullptr, nullptr, mb, 2048, 4096, 1024);
    // x = h2 + mlp @ fc2
    gemm_bf16_kernel<true,false,true,false><<<dim3(8, 16), blk, 0, stream>>>(
        mb, w2, h2f, x, nullptr, 2048, 1024, 4096);
  }

  // final LN + lm_head
  ln_kernel<<<2048, blk, 0, stream>>>(x, osc, nullptr, xfb);
  gemm_bf16_kernel<false,false,true,false><<<dim3(250, 16), blk, 0, stream>>>(
      xfb, lmh, nullptr, out, nullptr, 2048, 32000, 1024);
}
